// SparseCausalSelfAttention_48077863911930
// MI455X (gfx1250) — matrix operand order repaired and checked by reading, not compiled, not run on hardware
//
#include <hip/hip_runtime.h>
#include <hip/hip_bf16.h>

// ---------------------------------------------------------------------------
// Sparse (strided, stride=2) causal self-attention for MI455X (gfx1250).
// B=8, T=2048, C=512, H=8, D=64.  bf16 WMMA (v_wmma_f32_16x16x32_bf16) with
// fp32 accumulation; softmax in fp32; flash-attention streaming over K/V.
// Round 2: software-pipelined GEMM k-loops (32x64 per wave), early V-fragment
// loads in attention, global_prefetch of next K tile.
// ---------------------------------------------------------------------------

#define BATCH 8
#define SEQ   2048
#define CH    512
#define HEADS 8
#define HD    64
#define MROWS (BATCH * SEQ)   // 16384

typedef __attribute__((ext_vector_type(16))) __bf16 bf16x16;
typedef __attribute__((ext_vector_type(8)))  float  f32x8;
typedef __attribute__((ext_vector_type(4)))  float  f32x4;

union Frag {
  bf16x16 bf;
  f32x4   q[2];   // two 16-byte chunks
};

static __device__ __forceinline__ f32x8 wmma_bf16(const Frag& a, const Frag& b, f32x8 c) {
  return __builtin_amdgcn_wmma_f32_16x16x32_bf16(false, a.bf, false, b.bf,
                                                 (short)0, c, false, false);
}

static __device__ __forceinline__ float red_max16(float v) {
#pragma unroll
  for (int m = 1; m < 16; m <<= 1) v = fmaxf(v, __shfl_xor(v, m, 32));
  return v;
}
static __device__ __forceinline__ float red_sum16(float v) {
#pragma unroll
  for (int m = 1; m < 16; m <<= 1) v += __shfl_xor(v, m, 32);
  return v;
}

// ------------------------- converters --------------------------------------

__global__ void cvt_f32_to_bf16(const float* __restrict__ in,
                                __bf16* __restrict__ out, int n) {
  int i = blockIdx.x * blockDim.x + threadIdx.x;
  if (i < n) out[i] = (__bf16)in[i];
}

// W [K, N] fp32  ->  Wt [N, K] bf16   (K-contiguous per output column)
__global__ void cvt_transpose_bf16(const float* __restrict__ in,
                                   __bf16* __restrict__ out, int K, int N) {
  int i = blockIdx.x * blockDim.x + threadIdx.x;
  if (i < K * N) {
    int k = i / N, n = i % N;
    out[(size_t)n * K + k] = (__bf16)in[i];
  }
}

// ------------------------- 32x64 pipelined GEMM core -----------------------
// One wave computes a 32(M) x 64(N) tile over K=512, bf16 WMMA, double-
// buffered fragments so loads for k+32 are in flight while k's WMMAs issue.

static __device__ __forceinline__ void gemm_tile_32x64(
    const __bf16* __restrict__ A,   // [M, 512] row-major bf16
    const __bf16* __restrict__ Bt,  // [N, 512] row-major bf16 (B transposed)
    int mbase, int nbase, int h16, int l, f32x8 acc[2][4]) {

  const __bf16* arow0 = A + (size_t)(mbase + l) * CH;
  const __bf16* arow1 = A + (size_t)(mbase + 16 + l) * CH;
  const __bf16* brow0 = Bt + (size_t)(nbase +  0 + l) * CH + 16 * h16;
  const __bf16* brow1 = Bt + (size_t)(nbase + 16 + l) * CH + 16 * h16;
  const __bf16* brow2 = Bt + (size_t)(nbase + 32 + l) * CH + 16 * h16;
  const __bf16* brow3 = Bt + (size_t)(nbase + 48 + l) * CH + 16 * h16;

  Frag a0[2], a1[2], b[2][4];

  auto load_slot = [&](int s, int kk) {
    a0[s].q[0] = *(const f32x4*)(arow0 + kk + 16 * h16);
    a0[s].q[1] = *(const f32x4*)(arow0 + kk + 16 * h16 + 8);
    a1[s].q[0] = *(const f32x4*)(arow1 + kk + 16 * h16);
    a1[s].q[1] = *(const f32x4*)(arow1 + kk + 16 * h16 + 8);
    b[s][0].q[0] = *(const f32x4*)(brow0 + kk);
    b[s][0].q[1] = *(const f32x4*)(brow0 + kk + 8);
    b[s][1].q[0] = *(const f32x4*)(brow1 + kk);
    b[s][1].q[1] = *(const f32x4*)(brow1 + kk + 8);
    b[s][2].q[0] = *(const f32x4*)(brow2 + kk);
    b[s][2].q[1] = *(const f32x4*)(brow2 + kk + 8);
    b[s][3].q[0] = *(const f32x4*)(brow3 + kk);
    b[s][3].q[1] = *(const f32x4*)(brow3 + kk + 8);
  };

  load_slot(0, 0);
#pragma unroll
  for (int kk = 0; kk < CH; kk += 32) {
    const int cur = (kk >> 5) & 1;
    if (kk + 32 < CH) load_slot(cur ^ 1, kk + 32);   // prefetch next k-step
#pragma unroll
    for (int c = 0; c < 4; ++c) acc[0][c] = wmma_bf16(a0[cur], b[cur][c], acc[0][c]);
#pragma unroll
    for (int c = 0; c < 4; ++c) acc[1][c] = wmma_bf16(a1[cur], b[cur][c], acc[1][c]);
  }
}

// ------------------------- QKV GEMM ----------------------------------------
// Y = X[16384,512] @ W_qkv[512,1536]; scatter into Q (scaled by 1/sqrt(D))
// [B,H,T,D], K [B,H,T,D], Vt [B,H,D,T] (all bf16).

__global__ void gemm_qkv(const __bf16* __restrict__ A,   // [16384,512]
                         const __bf16* __restrict__ Bt,  // [1536,512]
                         __bf16* __restrict__ Qo,
                         __bf16* __restrict__ Ko,
                         __bf16* __restrict__ Vto) {
  const int wid  = threadIdx.x >> 5;
  const int lane = threadIdx.x & 31;
  const int h16  = lane >> 4;
  const int l    = lane & 15;
  const int mbase = (blockIdx.y * 4 + wid) * 32;
  const int nbase = blockIdx.x * 64;

  f32x8 acc[2][4] = {};
  gemm_tile_32x64(A, Bt, mbase, nbase, h16, l, acc);

  const int part = nbase >> 9;            // 0=q,1=k,2=v (uniform per block)
  const int head = (nbase & 511) >> 6;    // uniform per block
#pragma unroll
  for (int sub = 0; sub < 2; ++sub) {
#pragma unroll
    for (int c = 0; c < 4; ++c) {
#pragma unroll
      for (int j = 0; j < 8; ++j) {
        float v = acc[sub][c][j];
        int m = mbase + sub * 16 + j + 8 * h16;
        int b = m >> 11, t = m & (SEQ - 1);
        int d = c * 16 + l;
        size_t bh = (size_t)(b * HEADS + head);
        if (part == 0)      Qo[(bh * SEQ + t) * HD + d] = (__bf16)(v * 0.125f);
        else if (part == 1) Ko[(bh * SEQ + t) * HD + d] = (__bf16)v;
        else                Vto[(bh * HD + d) * SEQ + t] = (__bf16)v;
      }
    }
  }
}

// ------------------------- flash attention ---------------------------------
// One wave per (b, h, 16-row query tile). Streams 32 key columns per step:
// 4 WMMA for Q@K^T, fp32 online softmax, P staged via LDS, 4 WMMA for P@V.
// V-fragments are loaded before the softmax so their latency hides under the
// VALU work; next K tile is prefetched with global_prefetch_b8.

__global__ void attn_strided(const __bf16* __restrict__ Q,   // [B,H,T,D] (pre-scaled)
                             const __bf16* __restrict__ K,   // [B,H,T,D]
                             const __bf16* __restrict__ Vt,  // [B,H,D,T]
                             __bf16* __restrict__ att) {     // [B*T, C]
  __shared__ __align__(16) __bf16 lds[4][16][32];
  const int wid  = threadIdx.x >> 5;
  const int lane = threadIdx.x & 31;
  const int h16  = lane >> 4;
  const int l    = lane & 15;

  const int tile  = blockIdx.x * 4 + wid;      // 8192 tiles total
  const int mtile = tile & (SEQ / 16 - 1);     // 0..127
  const int bh    = tile >> 7;                 // 0..63
  const int head  = bh & (HEADS - 1);
  const int b     = bh >> 3;
  const int mbase = mtile * 16;

  const __bf16* q  = Q  + (size_t)bh * SEQ * HD;
  const __bf16* kp = K  + (size_t)bh * SEQ * HD;
  const __bf16* vt = Vt + (size_t)bh * HD * SEQ;

  // Q fragments for d-steps 0 and 32 (loaded once)
  Frag qf[2];
  {
    const __bf16* qrow = q + (size_t)(mbase + l) * HD;
    qf[0].q[0] = *(const f32x4*)(qrow + 16 * h16);
    qf[0].q[1] = *(const f32x4*)(qrow + 8 + 16 * h16);
    qf[1].q[0] = *(const f32x4*)(qrow + 32 + 16 * h16);
    qf[1].q[1] = *(const f32x4*)(qrow + 40 + 16 * h16);
  }

  f32x8 oacc[4] = {};
  float mrun[8], lrun[8];
#pragma unroll
  for (int j = 0; j < 8; ++j) { mrun[j] = -__builtin_inff(); lrun[j] = 0.f; }

  const int nsteps = ((mbase + 15) >> 5) + 1;  // causal: only tiles <= diagonal
  for (int ss = 0; ss < nsteps; ++ss) {
    const int sbase = ss * 32;

    // --- load K fragments (8 x b128), then V fragments (8 x b128) ---
    Frag k00, k01, k10, k11;
    {
      const __bf16* krow0 = kp + (size_t)(sbase + l) * HD + 16 * h16;
      const __bf16* krow1 = kp + (size_t)(sbase + 16 + l) * HD + 16 * h16;
      k00.q[0] = *(const f32x4*)(krow0);      k00.q[1] = *(const f32x4*)(krow0 + 8);
      k01.q[0] = *(const f32x4*)(krow0 + 32); k01.q[1] = *(const f32x4*)(krow0 + 40);
      k10.q[0] = *(const f32x4*)(krow1);      k10.q[1] = *(const f32x4*)(krow1 + 8);
      k11.q[0] = *(const f32x4*)(krow1 + 32); k11.q[1] = *(const f32x4*)(krow1 + 40);
    }
    Frag vf[4];
#pragma unroll
    for (int c = 0; c < 4; ++c) {
      const __bf16* vrow = vt + (size_t)(c * 16 + l) * SEQ + sbase + 16 * h16;
      vf[c].q[0] = *(const f32x4*)(vrow);
      vf[c].q[1] = *(const f32x4*)(vrow + 8);
    }
    // prefetch next step's K tile (32 rows, one line per lane); uniform branch
    if (ss + 1 < nsteps)
      __builtin_prefetch(kp + (size_t)(sbase + 32 + lane) * HD, 0, 1);

    // --- S[16x32] = Q @ K^T (needs only K fragments -> partial loadcnt wait)
    f32x8 s0 = {}, s1 = {};
    s0 = wmma_bf16(qf[0], k00, s0);
    s0 = wmma_bf16(qf[1], k01, s0);
    s1 = wmma_bf16(qf[0], k10, s1);
    s1 = wmma_bf16(qf[1], k11, s1);

    // --- mask + online softmax (fp32), write P (bf16) to LDS ---
#pragma unroll
    for (int j = 0; j < 8; ++j) {
      const int r   = j + 8 * h16;
      const int row = mbase + r;
      const int c0  = sbase + l;
      const int c1  = sbase + 16 + l;
      float v0 = s0[j], v1 = s1[j];
      bool u0 = (c0 == 0) | (c0 == row) | ((c0 < row) & (((row - 1 - c0) & 1) == 0));
      bool u1 = (c1 == 0) | (c1 == row) | ((c1 < row) & (((row - 1 - c1) & 1) == 0));
      v0 = u0 ? v0 : -1.0e30f;
      v1 = u1 ? v1 : -1.0e30f;
      float mv = red_max16(fmaxf(v0, v1));
      float mn = fmaxf(mrun[j], mv);
      float fs = __expf(mrun[j] - mn);        // 0 on first tile (mrun = -inf)
      float p0 = __expf(v0 - mn);
      float p1 = __expf(v1 - mn);
      float rs = red_sum16(p0 + p1);
      lrun[j] = lrun[j] * fs + rs;
      mrun[j] = mn;
      oacc[0][j] *= fs; oacc[1][j] *= fs; oacc[2][j] *= fs; oacc[3][j] *= fs;
      lds[wid][r][l]      = (__bf16)p0;
      lds[wid][r][16 + l] = (__bf16)p1;
    }

    __builtin_amdgcn_wave_barrier();
    asm volatile("s_wait_dscnt 0" ::: "memory");   // LDS is in-order per wave

    // --- O += P @ V (A-frag from LDS; V fragments already in flight) ---
    Frag pf;
    pf.q[0] = *(const f32x4*)(&lds[wid][l][16 * h16]);
    pf.q[1] = *(const f32x4*)(&lds[wid][l][16 * h16 + 8]);
#pragma unroll
    for (int c = 0; c < 4; ++c)
      oacc[c] = wmma_bf16(pf, vf[c], oacc[c]);
    __builtin_amdgcn_wave_barrier();               // keep next stores after loads
  }

  // --- normalize and store to att [B*T, C] as bf16 ---
#pragma unroll
  for (int j = 0; j < 8; ++j) {
    const int r = j + 8 * h16;
    const float inv = 1.0f / lrun[j];
    const size_t gm = (size_t)b * SEQ + mbase + r;
#pragma unroll
    for (int c = 0; c < 4; ++c)
      att[gm * CH + head * HD + c * 16 + l] = (__bf16)(oacc[c][j] * inv);
  }
}

// ------------------------- output projection -------------------------------
// out[16384,512] (fp32) = att[16384,512](bf16) @ W_proj (via Wp_t[512,512] bf16)

__global__ void gemm_proj(const __bf16* __restrict__ A,
                          const __bf16* __restrict__ Bt,
                          float* __restrict__ out) {
  const int wid  = threadIdx.x >> 5;
  const int lane = threadIdx.x & 31;
  const int h16  = lane >> 4;
  const int l    = lane & 15;
  const int mbase = (blockIdx.y * 4 + wid) * 32;
  const int nbase = blockIdx.x * 64;

  f32x8 acc[2][4] = {};
  gemm_tile_32x64(A, Bt, mbase, nbase, h16, l, acc);

#pragma unroll
  for (int sub = 0; sub < 2; ++sub) {
#pragma unroll
    for (int c = 0; c < 4; ++c) {
#pragma unroll
      for (int j = 0; j < 8; ++j) {
        int m = mbase + sub * 16 + j + 8 * h16;
        out[(size_t)m * CH + nbase + c * 16 + l] = acc[sub][c][j];
      }
    }
  }
}

// ------------------------- launcher ----------------------------------------

extern "C" void kernel_launch(void* const* d_in, const int* in_sizes, int n_in,
                              void* d_out, int out_size, void* d_ws, size_t ws_size,
                              hipStream_t stream) {
  (void)in_sizes; (void)n_in; (void)out_size; (void)ws_size;
  const float* x      = (const float*)d_in[0];   // [8,2048,512]
  const float* W_qkv  = (const float*)d_in[1];   // [512,1536]
  const float* W_proj = (const float*)d_in[2];   // [512,512]
  float* out = (float*)d_out;                    // [8,2048,512]

  // workspace carve (bf16 buffers), ~82 MiB total
  char* p = (char*)d_ws;
  const size_t SZ_X   = (size_t)MROWS * CH * 2;                 // 16 MiB
  const size_t SZ_HT  = (size_t)BATCH * HEADS * SEQ * HD * 2;   // 16 MiB
  const size_t SZ_WQ  = (size_t)CH * 3 * CH * 2;                // 1.5 MiB
  const size_t SZ_WP  = (size_t)CH * CH * 2;                    // 0.5 MiB
  __bf16* xb  = (__bf16*)p;            p += SZ_X;
  __bf16* qb  = (__bf16*)p;            p += SZ_HT;
  __bf16* kb  = (__bf16*)p;            p += SZ_HT;
  __bf16* vtb = (__bf16*)p;            p += SZ_HT;
  __bf16* atb = (__bf16*)p;            p += SZ_X;
  __bf16* wqt = (__bf16*)p;            p += SZ_WQ;
  __bf16* wpt = (__bf16*)p;            p += SZ_WP;

  const int nX = MROWS * CH;
  cvt_f32_to_bf16<<<(nX + 255) / 256, 256, 0, stream>>>(x, xb, nX);
  cvt_transpose_bf16<<<(CH * 3 * CH + 255) / 256, 256, 0, stream>>>(W_qkv, wqt, CH, 3 * CH);
  cvt_transpose_bf16<<<(CH * CH + 255) / 256, 256, 0, stream>>>(W_proj, wpt, CH, CH);

  // QKV GEMM: M=16384 (y: 16384/128=128), N=1536 (x: 1536/64=24)
  gemm_qkv<<<dim3(24, 128), 128, 0, stream>>>(xb, wqt, qb, kb, vtb);

  // attention: B*H*(T/16) = 8192 waves -> 2048 blocks of 4 waves
  attn_strided<<<2048, 128, 0, stream>>>(qb, kb, vtb, atb);

  // projection: N=512 (x: 8), M=16384 (y: 128)
  gemm_proj<<<dim3(8, 128), 128, 0, stream>>>(atb, wpt, out);
}
